// DPXEmbedder_50629074485724
// MI455X (gfx1250) — matrix operand order repaired and checked by reading, not compiled, not run on hardware
//
#include <hip/hip_runtime.h>
#include <hip/hip_bf16.h>

typedef __bf16 bf16_t;
typedef __attribute__((ext_vector_type(16))) __bf16 v16bf;
typedef __attribute__((ext_vector_type(8)))  float  v8f;

#define BDIM 8
#define TDIM 1024
#define DDIM 768
#define NCLS 1
#define BM 128
#define BN 64
#define BK 32

// ---------- conversion kernels ----------
__global__ __launch_bounds__(256) void conv_fv_kernel(const float* __restrict__ fV,
                                                      bf16_t* __restrict__ fVb,
                                                      long long total) {
    long long i = (long long)blockIdx.x * blockDim.x + threadIdx.x;
    if (i < total) fVb[i] = (bf16_t)fV[i];
}

__global__ __launch_bounds__(256) void conv_wt_kernel(const float* __restrict__ W,
                                                      bf16_t* __restrict__ Wt) {
    int i = blockIdx.x * blockDim.x + threadIdx.x;  // over T*D
    if (i < TDIM * DDIM) {
        int t = i / DDIM, d = i % DDIM;
        Wt[(long long)d * TDIM + t] = (bf16_t)W[i];
    }
}

// ---------- segment -> batch ownership ----------
__global__ __launch_bounds__(256) void seg_batch_kernel(const int* __restrict__ seg,
                                                        const int* __restrict__ byx0,
                                                        int* __restrict__ batch_of,
                                                        int total) {
    int i = blockIdx.x * blockDim.x + threadIdx.x;
    if (i < total) batch_of[seg[i]] = byx0[i];   // same value per id; race benign
}

__global__ __launch_bounds__(256) void counts_kernel(const int* __restrict__ batch_of,
                                                     int* __restrict__ bc,
                                                     int* __restrict__ st,
                                                     int nrows) {
    __shared__ int cnt[BDIM];
    int tid = threadIdx.x;
    if (tid < BDIM) cnt[tid] = 0;
    __syncthreads();
    for (int i = tid; i < nrows; i += 256) atomicAdd(&cnt[batch_of[i]], 1);
    __syncthreads();
    if (tid == 0) {
        int acc = 0;
        for (int b = 0; b < BDIM; ++b) { bc[b] = cnt[b]; st[b] = acc; acc += cnt[b]; }
        st[BDIM] = acc;
    }
}

// ---------- WMMA GEMM with async global->LDS double buffering ----------
__global__ __launch_bounds__(256) void gemm_wmma_kernel(const bf16_t* __restrict__ A,   // [nrows, T] bf16
                                                        const bf16_t* __restrict__ Bt,  // [D, T] bf16 (W^T)
                                                        const float* __restrict__ bias, // [D]
                                                        const int* __restrict__ st,     // [B+1]
                                                        float* __restrict__ tok,        // [B, maxdim, D]
                                                        int nrows, int maxdim) {
    __shared__ __align__(16) bf16_t Alds[2][BM * BK];  // 2 x 8 KB, k-contiguous
    __shared__ __align__(16) bf16_t Blds[2][BN * BK];  // 2 x 4 KB, n-major k-contiguous

    const int tid  = threadIdx.x;
    const int wave = tid >> 5;
    const int lane = tid & 31;
    const int ntiles = DDIM / BN;                 // 12
    const int row0 = (blockIdx.x / ntiles) * BM;
    const int col0 = (blockIdx.x % ntiles) * BN;
    const int wm = wave >> 1;                     // 0..3
    const int wn = wave & 1;                      // 0..1

    // per-thread async-copy slots (3 x b128 per thread per K-step)
    const int ar0 = tid >> 2;                     // A row, chunk 0 (0..63)
    const int ar1 = 64 + (tid >> 2);              // A row, chunk 1 (64..127)
    const int ac  = (tid & 3) * 8;                // k-offset within row
    int garow0 = row0 + ar0; if (garow0 >= nrows) garow0 = nrows - 1;
    int garow1 = row0 + ar1; if (garow1 >= nrows) garow1 = nrows - 1;
    const int br = tid >> 2;                      // B n-row (0..63)

    v8f acc[2][2] = {};

    const int mrow  = lane & 15;
    const int koffA = (lane < 16) ? 0 : 8;
    const int koffB = (lane < 16) ? 0 : 8;

#define ISSUE_TILE(buf, k0)                                                              \
    do {                                                                                 \
        unsigned l0 = (unsigned)(size_t)&Alds[(buf)][ar0 * BK + ac];                     \
        unsigned long long g0 =                                                          \
            (unsigned long long)(size_t)&A[(long long)garow0 * TDIM + (k0) + ac];        \
        asm volatile("global_load_async_to_lds_b128 %0, %1, off" ::"v"(l0), "v"(g0)      \
                     : "memory");                                                        \
        unsigned l1 = (unsigned)(size_t)&Alds[(buf)][ar1 * BK + ac];                     \
        unsigned long long g1 =                                                          \
            (unsigned long long)(size_t)&A[(long long)garow1 * TDIM + (k0) + ac];        \
        asm volatile("global_load_async_to_lds_b128 %0, %1, off" ::"v"(l1), "v"(g1)      \
                     : "memory");                                                        \
        unsigned l2 = (unsigned)(size_t)&Blds[(buf)][br * BK + ac];                      \
        unsigned long long g2 =                                                          \
            (unsigned long long)(size_t)&Bt[(long long)(col0 + br) * TDIM + (k0) + ac];  \
        asm volatile("global_load_async_to_lds_b128 %0, %1, off" ::"v"(l2), "v"(g2)      \
                     : "memory");                                                        \
    } while (0)

    ISSUE_TILE(0, 0);

    int buf = 0;
    for (int k0 = 0; k0 < TDIM; k0 += BK, buf ^= 1) {
        // our async copies for the current buffer have landed in LDS
        asm volatile("s_wait_asynccnt 0" ::: "memory");
        // everyone's copies landed; everyone's reads of the other buffer retired
        __syncthreads();
        // prefetch next K-step into the other buffer while we compute
        if (k0 + BK < TDIM) ISSUE_TILE(buf ^ 1, k0 + BK);

        union Frag { v16bf v; uint4 u[2]; };
        Frag a[2], bb[2];
#pragma unroll
        for (int mi = 0; mi < 2; ++mi) {
            const bf16_t* ap = &Alds[buf][(wm * 32 + mi * 16 + mrow) * BK + koffA];
            a[mi].u[0] = *(const uint4*)(ap);       // K lo 8
            a[mi].u[1] = *(const uint4*)(ap + 16);  // K hi 8
        }
#pragma unroll
        for (int ni = 0; ni < 2; ++ni) {
            const bf16_t* bp = &Blds[buf][(wn * 32 + ni * 16 + mrow) * BK + koffB];
            bb[ni].u[0] = *(const uint4*)(bp);      // K lo 8
            bb[ni].u[1] = *(const uint4*)(bp + 16); // K hi 8
        }
#pragma unroll
        for (int mi = 0; mi < 2; ++mi)
#pragma unroll
            for (int ni = 0; ni < 2; ++ni)
                acc[mi][ni] = __builtin_amdgcn_wmma_f32_16x16x32_bf16(
                    false, a[mi].v, false, bb[ni].v, (short)0, acc[mi][ni], false, false);
        __syncthreads();   // all reads of this buffer done before it is refilled
    }
#undef ISSUE_TILE

    // epilogue: bias add + scatter rows straight into tok[b, j, :]
    int stv[BDIM + 1];
#pragma unroll
    for (int i = 0; i <= BDIM; ++i) stv[i] = st[i];

    const int colBase = col0 + wn * 32 + (lane & 15);
    const int rowHalf = (lane < 16) ? 0 : 8;
#pragma unroll
    for (int mi = 0; mi < 2; ++mi) {
#pragma unroll
        for (int ni = 0; ni < 2; ++ni) {
            int col = colBase + ni * 16;
            float bv = bias[col];
#pragma unroll
            for (int r = 0; r < 8; ++r) {
                int grow = row0 + wm * 32 + mi * 16 + rowHalf + r;
                if (grow < nrows) {
                    int b = 0;
#pragma unroll
                    for (int q = 1; q < BDIM; ++q) if (grow >= stv[q]) b = q;
                    int j = grow - stv[b] + NCLS;
                    tok[((long long)(b * maxdim + j)) * DDIM + col] = acc[mi][ni][r] + bv;
                }
            }
        }
    }
}

// ---------- cls row + zero padding ----------
__global__ __launch_bounds__(256) void fill_kernel(float* __restrict__ tok,
                                                   const float* __restrict__ cls,
                                                   const int* __restrict__ bc,
                                                   int maxdim) {
    int rowid = blockIdx.x;
    int b = rowid / maxdim, j = rowid % maxdim;
    int tid = threadIdx.x;
    float* rowp = tok + (long long)rowid * DDIM;
    if (j < NCLS) {
#pragma unroll
        for (int i = 0; i < 3; ++i) rowp[tid + i * 256] = cls[j * DDIM + tid + i * 256];
    } else if (j - NCLS >= bc[b]) {
#pragma unroll
        for (int i = 0; i < 3; ++i) rowp[tid + i * 256] = 0.0f;
    }
}

// ---------- in-place layernorm + amask ----------
__global__ __launch_bounds__(256) void ln_kernel(float* __restrict__ tok,
                                                 const float* __restrict__ gamma,
                                                 const float* __restrict__ beta,
                                                 const int* __restrict__ bc,
                                                 float* __restrict__ amask,
                                                 int maxdim) {
    int rowid = blockIdx.x;
    int b = rowid / maxdim, j = rowid % maxdim;
    int tid = threadIdx.x;
    float* rowp = tok + (long long)rowid * DDIM;

    float x[3], s = 0.f, ss = 0.f;
#pragma unroll
    for (int i = 0; i < 3; ++i) {
        x[i] = rowp[tid + i * 256];
        s += x[i];
        ss += x[i] * x[i];
    }
    __shared__ float sh1[256], sh2[256];
    sh1[tid] = s; sh2[tid] = ss;
    __syncthreads();
    for (int off = 128; off >= 1; off >>= 1) {
        if (tid < off) { sh1[tid] += sh1[tid + off]; sh2[tid] += sh2[tid + off]; }
        __syncthreads();
    }
    float mu  = sh1[0] * (1.0f / DDIM);
    float var = sh2[0] * (1.0f / DDIM) - mu * mu;
    float rs  = rsqrtf(var + 1e-5f);
#pragma unroll
    for (int i = 0; i < 3; ++i) {
        int c = tid + i * 256;
        rowp[c] = (x[i] - mu) * rs * gamma[c] + beta[c];
    }
    if (tid == 0) amask[rowid] = (j < bc[b] + NCLS) ? 1.0f : 0.0f;
}

// ---------- seg -> float copy ----------
__global__ __launch_bounds__(256) void seg_copy_kernel(const int* __restrict__ seg,
                                                       float* __restrict__ out, int total) {
    int i = blockIdx.x * blockDim.x + threadIdx.x;
    if (i < total) out[i] = (float)seg[i];
}

extern "C" void kernel_launch(void* const* d_in, const int* in_sizes, int n_in,
                              void* d_out, int out_size, void* d_ws, size_t ws_size,
                              hipStream_t stream) {
    const float* fV   = (const float*)d_in[0];
    const int*   seg  = (const int*)d_in[1];
    const int*   byx0 = (const int*)d_in[2];
    const float* W    = (const float*)d_in[3];
    const float* bias = (const float*)d_in[4];
    const float* cls  = (const float*)d_in[5];
    const float* gam  = (const float*)d_in[6];
    const float* bet  = (const float*)d_in[7];

    const int nrows    = in_sizes[0] / TDIM;                          // 30500
    const int segTotal = in_sizes[1];                                 // B*H*W
    const int maxdim   = (out_size - segTotal) / (BDIM * (DDIM + 1)); // 4097

    float* out_tok   = (float*)d_out;
    float* out_seg   = out_tok + (long long)BDIM * maxdim * DDIM;
    float* out_amask = out_seg + segTotal;

    // workspace carve-up
    char* ws = (char*)d_ws;
    size_t off = 0;
    bf16_t* fVb = (bf16_t*)(ws + off); off += (size_t)nrows * TDIM * sizeof(bf16_t); off = (off + 255) & ~(size_t)255;
    bf16_t* Wt  = (bf16_t*)(ws + off); off += (size_t)TDIM * DDIM * sizeof(bf16_t); off = (off + 255) & ~(size_t)255;
    int* batch_of = (int*)(ws + off);  off += (size_t)nrows * sizeof(int);          off = (off + 255) & ~(size_t)255;
    int* bc = (int*)(ws + off);        off += 256;
    int* st = (int*)(ws + off);        off += 256;

    // 1) conversions
    long long fvTotal = (long long)nrows * TDIM;
    conv_fv_kernel<<<(int)((fvTotal + 255) / 256), 256, 0, stream>>>(fV, fVb, fvTotal);
    conv_wt_kernel<<<(TDIM * DDIM + 255) / 256, 256, 0, stream>>>(W, Wt);

    // 2) segment ownership -> counts/offsets
    seg_batch_kernel<<<(segTotal + 255) / 256, 256, 0, stream>>>(seg, byx0, batch_of, segTotal);
    counts_kernel<<<1, 256, 0, stream>>>(batch_of, bc, st, nrows);

    // 3) WMMA GEMM with async-LDS double buffering, fused bias + ragged scatter
    int mtiles = (nrows + BM - 1) / BM;
    gemm_wmma_kernel<<<mtiles * (DDIM / BN), 256, 0, stream>>>(fVb, Wt, bias, st, out_tok, nrows, maxdim);

    // 4) cls + zero padding, then in-place layernorm + amask
    fill_kernel<<<BDIM * maxdim, 256, 0, stream>>>(out_tok, cls, bc, maxdim);
    ln_kernel<<<BDIM * maxdim, 256, 0, stream>>>(out_tok, gam, bet, bc, out_amask, maxdim);

    // 5) seg passthrough (as float)
    seg_copy_kernel<<<(segTotal + 255) / 256, 256, 0, stream>>>(seg, out_seg, segTotal);
}
